// Dual_QK_mutlihead_attn_65712999629510
// MI455X (gfx1250) — matrix operand order repaired and checked by reading, not compiled, not run on hardware
//
#include <hip/hip_runtime.h>
#include <hip/hip_bf16.h>

// ---------------------------------------------------------------------------
// Dual-QK multi-head attention, CDNA5 (gfx1250), bf16 WMMA pipeline.
// B=2, S=2048, HID=1024, NH=16, HD=64.  M = B*S = 4096 rows.
// ---------------------------------------------------------------------------

typedef unsigned short u16;
typedef __attribute__((ext_vector_type(16))) __bf16         v16bf;
typedef __attribute__((ext_vector_type(8)))  float          v8f;
typedef __attribute__((ext_vector_type(8)))  unsigned short v8u16;

union Frag {                 // one WMMA 16-bit operand (16 elements / lane)
    v8u16 h[2];
    v16bf v;
};

__device__ __forceinline__ u16 cvt_bf16(float f) {
    unsigned int u = __float_as_uint(f);
    u += 0x7fffu + ((u >> 16) & 1u);          // round-to-nearest-even
    return (u16)(u >> 16);
}

__device__ __forceinline__ v8f vzero8() {
    v8f z = {0.f, 0.f, 0.f, 0.f, 0.f, 0.f, 0.f, 0.f};
    return z;
}

#define WMMA_BF16(A, Bv, C) \
    __builtin_amdgcn_wmma_f32_16x16x32_bf16(false, (A), false, (Bv), (short)0, (C), false, false)

// ---------------------------------------------------------------------------
// Async HBM -> LDS copy (CDNA5 GLOBAL_LOAD_ASYNC_TO_LDS_B128, ASYNCcnt).
// Toolchain declares it as (i32x4 AS(1)*, i32x4 AS(3)*, imm offset, imm cpol).
// Guarded: falls back to a synchronous copy if the builtins are not declared.
// ---------------------------------------------------------------------------
#if defined(__AMDGCN__) && __has_builtin(__builtin_amdgcn_global_load_async_to_lds_b128) && \
    __has_builtin(__builtin_amdgcn_s_wait_asynccnt)
#define USE_ASYNC_LDS 1
typedef int i32x4 __attribute__((vector_size(16)));
typedef __attribute__((address_space(1))) void*  as1_vp;
typedef __attribute__((address_space(3))) void*  as3_vp;
typedef __attribute__((address_space(1))) i32x4* as1_i32x4p;
typedef __attribute__((address_space(3))) i32x4* as3_i32x4p;
__device__ __forceinline__ void cp_async_b128(const u16* g, u16* l) {
    __builtin_amdgcn_global_load_async_to_lds_b128(
        (as1_i32x4p)(as1_vp)(void*)g,
        (as3_i32x4p)(as3_vp)(void*)l,
        0, 0);
}
#define ASYNC_WAIT(n) __builtin_amdgcn_s_wait_asynccnt(n)
#else
__device__ __forceinline__ void cp_async_b128(const u16* g, u16* l) {
    *(v8u16*)l = *(const v8u16*)g;
}
#define ASYNC_WAIT(n) ((void)0)
#endif

// ---------------------------------------------------------------------------
// Kernel 1: fp32 -> bf16 elementwise (weights).  n divisible by 1024.
// ---------------------------------------------------------------------------
__global__ void __launch_bounds__(256)
k_cvt_bf16(const float* __restrict__ src, u16* __restrict__ dst, int n)
{
    int i = (blockIdx.x * 256 + threadIdx.x) * 4;
    if (i >= n) return;
    float4 v = *(const float4*)(src + i);
    unsigned int lo = (unsigned int)cvt_bf16(v.x) | ((unsigned int)cvt_bf16(v.y) << 16);
    unsigned int hi = (unsigned int)cvt_bf16(v.z) | ((unsigned int)cvt_bf16(v.w) << 16);
    uint2 pk; pk.x = lo; pk.y = hi;
    *(uint2*)&dst[i] = pk;
}

// ---------------------------------------------------------------------------
// Kernel 2: row LayerNorm (optionally of x+x2) over 1024 cols -> bf16.
// ---------------------------------------------------------------------------
__global__ void __launch_bounds__(256)
k_layernorm_bf16(const float* __restrict__ x, const float* __restrict__ x2,
                 const float* __restrict__ g, const float* __restrict__ bta,
                 u16* __restrict__ out)
{
    const int row = blockIdx.x;
    const int tid = threadIdx.x;
    const int col = tid * 4;
    const float* xr = x + (size_t)row * 1024;

    float4 v = *(const float4*)(xr + col);
    if (x2) {
        float4 w = *(const float4*)(x2 + (size_t)row * 1024 + col);
        v.x += w.x; v.y += w.y; v.z += w.z; v.w += w.w;
    }
    float s1 = v.x + v.y + v.z + v.w;
    float s2 = v.x * v.x + v.y * v.y + v.z * v.z + v.w * v.w;

    #pragma unroll
    for (int o = 16; o > 0; o >>= 1) {
        s1 += __shfl_xor(s1, o, 32);
        s2 += __shfl_xor(s2, o, 32);
    }
    __shared__ float r1[8], r2[8];
    if ((tid & 31) == 0) { r1[tid >> 5] = s1; r2[tid >> 5] = s2; }
    __syncthreads();
    float ts = 0.f, tq = 0.f;
    #pragma unroll
    for (int i = 0; i < 8; ++i) { ts += r1[i]; tq += r2[i]; }

    const float mean = ts * (1.0f / 1024.0f);
    const float var  = tq * (1.0f / 1024.0f) - mean * mean;
    const float rstd = rsqrtf(var + 1e-12f);

    float4 gg = *(const float4*)(g + col);
    float4 bb = *(const float4*)(bta + col);
    float o0 = (v.x - mean) * rstd * gg.x + bb.x;
    float o1 = (v.y - mean) * rstd * gg.y + bb.y;
    float o2 = (v.z - mean) * rstd * gg.z + bb.z;
    float o3 = (v.w - mean) * rstd * gg.w + bb.w;

    unsigned int lo = (unsigned int)cvt_bf16(o0) | ((unsigned int)cvt_bf16(o1) << 16);
    unsigned int hi = (unsigned int)cvt_bf16(o2) | ((unsigned int)cvt_bf16(o3) << 16);
    uint2 pk; pk.x = lo; pk.y = hi;
    *(uint2*)&out[(size_t)row * 1024 + col] = pk;
}

// ---------------------------------------------------------------------------
// Kernel 3: Y[4096,1024] = X(bf16) @ W(bf16)^T + bias, WMMA bf16.
// Block tile 128x128, BK=64 (2 WMMA k-steps/stage), 8 waves (4x2),
// wave tile 32x64 (2x4 WMMA tiles).  Double-buffered LDS with async
// HBM->LDS prefetch of the next K-stage overlapped with current compute.
// OUTF32=0: store bf16 to outB.   OUTF32=1: store f32 (+resid) to outF.
// ---------------------------------------------------------------------------
template<int OUTF32>
__global__ void __launch_bounds__(256)
k_gemm_wmma(const u16* __restrict__ X, const u16* __restrict__ W,
            const float* __restrict__ bias, const float* __restrict__ resid,
            float* __restrict__ outF, u16* __restrict__ outB)
{
    constexpr int LDT = 72;                         // 64 + 8 pad (bf16 elems)
    __shared__ alignas(16) u16 lA[2][128 * LDT];
    __shared__ alignas(16) u16 lB[2][128 * LDT];

    const int tid  = threadIdx.x;
    const int wid  = tid >> 5;
    const int lane = tid & 31;
    const int hf   = lane >> 4;
    const int l    = lane & 15;
    const int bm   = blockIdx.y * 128;
    const int bn   = blockIdx.x * 128;
    const int wm   = (wid >> 1) * 32;               // wave M offset in tile
    const int wn   = (wid & 1) * 64;                // wave N offset in tile

    v8f acc[2][4];
    #pragma unroll
    for (int tm = 0; tm < 2; ++tm)
        #pragma unroll
        for (int tn = 0; tn < 4; ++tn) acc[tm][tn] = vzero8();

    auto load_stage = [&](int buf, int k0) {
        // 128 rows x 64 cols per matrix = 1024 16B chunks; 4 A + 4 B per thread
        for (int c = tid; c < 1024; c += 256) {
            const int row = c >> 3, sub = (c & 7) * 8;
            cp_async_b128(&X[(size_t)(bm + row) * 1024 + k0 + sub],
                          &lA[buf][row * LDT + sub]);
            cp_async_b128(&W[(size_t)(bn + row) * 1024 + k0 + sub],
                          &lB[buf][row * LDT + sub]);
        }
    };

    load_stage(0, 0);
    for (int it = 0; it < 16; ++it) {               // 1024 / 64 stages
        const int buf = it & 1;
        if (it + 1 < 16) {
            load_stage(buf ^ 1, (it + 1) * 64);
            ASYNC_WAIT(8);                          // stage `it` complete (in-order)
        } else {
            ASYNC_WAIT(0);
        }
        __syncthreads();

        #pragma unroll
        for (int ks = 0; ks < 2; ++ks) {            // two K=32 sub-steps
            Frag a[2], b[4];
            #pragma unroll
            for (int tm = 0; tm < 2; ++tm) {
                const u16* p = &lA[buf][(wm + tm * 16 + l) * LDT + ks * 32 + hf * 8];
                a[tm].h[0] = *(const v8u16*)p;          // K = hf*8 .. +7
                a[tm].h[1] = *(const v8u16*)(p + 16);   // K = 16+hf*8 .. +7
            }
            #pragma unroll
            for (int tn = 0; tn < 4; ++tn) {
                const u16* p = &lB[buf][(wn + tn * 16 + l) * LDT + ks * 32 + hf * 8];
                b[tn].h[0] = *(const v8u16*)p;          // K = hf*8 .. +7
                b[tn].h[1] = *(const v8u16*)(p + 16);   // K = 16+hf*8 .. +7
            }
            #pragma unroll
            for (int tm = 0; tm < 2; ++tm)
                #pragma unroll
                for (int tn = 0; tn < 4; ++tn)
                    acc[tm][tn] = WMMA_BF16(a[tm].v, b[tn].v, acc[tm][tn]);
        }
        __syncthreads();
    }

    #pragma unroll
    for (int tm = 0; tm < 2; ++tm) {
        #pragma unroll
        for (int tn = 0; tn < 4; ++tn) {
            #pragma unroll
            for (int j = 0; j < 8; ++j) {
                const int row = bm + wm + tm * 16 + (hf ? j + 8 : j);
                const int col = bn + wn + tn * 16 + l;
                float v = acc[tm][tn][j] + bias[col];
                if (OUTF32) {
                    outF[(size_t)row * 1024 + col] = v + resid[(size_t)row * 1024 + col];
                } else {
                    outB[(size_t)row * 1024 + col] = cvt_bf16(v);
                }
            }
        }
    }
}

// ---------------------------------------------------------------------------
// Kernel 4: flash attention for one stream.
// grid = (S/128, B*NH).  8 waves; wave owns 16 query rows.
// K tile [128][64] LDS row-major (async HBM->LDS); V tile transposed [64][128]
// LDS (sync transpose scatter); per-wave P scratch [16][128] bf16 for the
// C-layout -> A-layout crossover.  Streaming (flash) softmax.
// ---------------------------------------------------------------------------
__global__ void __launch_bounds__(256)
k_attention(const u16* __restrict__ Q, const u16* __restrict__ K,
            const u16* __restrict__ V, const float* __restrict__ mask,
            float* __restrict__ ctx)
{
    constexpr int LDK = 72;    // 64 + 8 pad
    constexpr int LDV = 136;   // 128 + 8 pad
    constexpr int LDP = 136;
    __shared__ alignas(16) u16  lK[128 * LDK];
    __shared__ alignas(16) u16  lVt[64 * LDV];
    __shared__ alignas(16) u16  lP[8 * 16 * LDP];
    __shared__ float lMask[128];

    const int tid  = threadIdx.x;
    const int wid  = tid >> 5;
    const int lane = tid & 31;
    const int hf   = lane >> 4;
    const int l    = lane & 15;
    const int bh   = blockIdx.y;
    const int b    = bh >> 4;
    const int h    = bh & 15;
    const int q0   = blockIdx.x * 128 + wid * 16;

    const u16* Qb = Q + (size_t)b * 2048 * 1024 + h * 64;
    const u16* Kb = K + (size_t)b * 2048 * 1024 + h * 64;
    const u16* Vb = V + (size_t)b * 2048 * 1024 + h * 64;

    // Q fragments for this wave's 16 rows (HD=64 -> two K=32 chunks), from HBM.
    Frag qa[2];
    {
        const u16* qr = Qb + (size_t)(q0 + l) * 1024;
        #pragma unroll
        for (int kc = 0; kc < 2; ++kc) {
            qa[kc].h[0] = *(const v8u16*)(qr + kc * 32 + hf * 8);
            qa[kc].h[1] = *(const v8u16*)(qr + kc * 32 + 16 + hf * 8);
        }
    }

    v8f ot[4];
    #pragma unroll
    for (int n = 0; n < 4; ++n) ot[n] = vzero8();
    float mrow[8], lrow[8];
    #pragma unroll
    for (int j = 0; j < 8; ++j) { mrow[j] = -3.0e38f; lrow[j] = 0.f; }
    const float scale = 0.125f;   // 1/sqrt(64)
    u16* myP = &lP[wid * 16 * LDP];

    for (int kv0 = 0; kv0 < 2048; kv0 += 128) {
        // K tile: async HBM->LDS (no VGPR round-trip)
        for (int c = tid; c < 1024; c += 256) {
            const int row = c >> 3, sub = (c & 7) * 8;
            cp_async_b128(Kb + (size_t)(kv0 + row) * 1024 + sub, &lK[row * LDK + sub]);
        }
        // V tile: load + transpose scatter (needs per-element placement)
        for (int c = tid; c < 1024; c += 256) {
            const int row = c >> 3, sub = (c & 7) * 8;
            v8u16 vv = *(const v8u16*)(Vb + (size_t)(kv0 + row) * 1024 + sub);
            #pragma unroll
            for (int e = 0; e < 8; ++e) lVt[(sub + e) * LDV + row] = vv[e];
        }
        if (tid < 128) lMask[tid] = mask[b * 2048 + kv0 + tid];
        ASYNC_WAIT(0);
        __syncthreads();

        // S = Q K^T  (8 tiles of 16 kv-cols, contraction over HD=64)
        v8f st[8];
        #pragma unroll
        for (int t = 0; t < 8; ++t) {
            v8f s = vzero8();
            #pragma unroll
            for (int kc = 0; kc < 2; ++kc) {
                Frag kb;
                const u16* p = &lK[(t * 16 + l) * LDK + kc * 32 + hf * 8];
                kb.h[0] = *(const v8u16*)p;
                kb.h[1] = *(const v8u16*)(p + 16);
                s = WMMA_BF16(qa[kc].v, kb.v, s);
            }
            st[t] = s;
        }

        float cmask[8];
        #pragma unroll
        for (int t = 0; t < 8; ++t) cmask[t] = lMask[t * 16 + l];

        // online softmax per row (row j / j+8 lives in a 16-lane half)
        #pragma unroll
        for (int j = 0; j < 8; ++j) {
            float mx = -3.0e38f;
            #pragma unroll
            for (int t = 0; t < 8; ++t) {
                float sv = st[t][j] * scale + cmask[t];
                mx = fmaxf(mx, sv);
            }
            #pragma unroll
            for (int o = 1; o < 16; o <<= 1) mx = fmaxf(mx, __shfl_xor(mx, o, 32));
            const float mnew = fmaxf(mrow[j], mx);
            const float f = __expf(mrow[j] - mnew);
            mrow[j] = mnew;
            lrow[j] *= f;
            #pragma unroll
            for (int n = 0; n < 4; ++n) ot[n][j] *= f;
            float rs = 0.f;
            #pragma unroll
            for (int t = 0; t < 8; ++t) {
                float sv = st[t][j] * scale + cmask[t];
                float pv = __expf(sv - mnew);
                rs += pv;
                myP[(hf ? j + 8 : j) * LDP + t * 16 + l] = cvt_bf16(pv);
            }
            #pragma unroll
            for (int o = 1; o < 16; o <<= 1) rs += __shfl_xor(rs, o, 32);
            lrow[j] += rs;
        }

        // O += P V  (contraction over 128 kv, 4 K=32 chunks; 4 d-tiles of 16)
        #pragma unroll
        for (int kc = 0; kc < 4; ++kc) {
            Frag pa;
            const u16* pp = &myP[l * LDP + kc * 32 + hf * 8];
            pa.h[0] = *(const v8u16*)pp;
            pa.h[1] = *(const v8u16*)(pp + 16);
            #pragma unroll
            for (int n = 0; n < 4; ++n) {
                Frag vb;
                const u16* vp = &lVt[(n * 16 + l) * LDV + kc * 32 + hf * 8];
                vb.h[0] = *(const v8u16*)vp;
                vb.h[1] = *(const v8u16*)(vp + 16);
                ot[n] = WMMA_BF16(pa.v, vb.v, ot[n]);
            }
        }
        __syncthreads();
    }

    #pragma unroll
    for (int n = 0; n < 4; ++n) {
        #pragma unroll
        for (int j = 0; j < 8; ++j) {
            const int row = q0 + (hf ? j + 8 : j);
            const int col = h * 64 + n * 16 + l;
            ctx[((size_t)(b * 2048 + row)) * 1024 + col] = ot[n][j] / lrow[j];
        }
    }
}

// ---------------------------------------------------------------------------
// Host-side orchestration.
// ---------------------------------------------------------------------------
extern "C" void kernel_launch(void* const* d_in, const int* in_sizes, int n_in,
                              void* d_out, int out_size, void* d_ws, size_t ws_size,
                              hipStream_t stream)
{
    (void)in_sizes; (void)n_in; (void)out_size; (void)ws_size;

    const float* x_q   = (const float*)d_in[0];
    const float* x_k   = (const float*)d_in[1];
    const float* x_q1  = (const float*)d_in[2];
    const float* x_k1  = (const float*)d_in[3];
    const float* x_v   = (const float*)d_in[4];
    const float* amask = (const float*)d_in[5];
    const float* Wq  = (const float*)d_in[6];   const float* bq  = (const float*)d_in[7];
    const float* Wk  = (const float*)d_in[8];   const float* bk  = (const float*)d_in[9];
    const float* Wq1 = (const float*)d_in[10];  const float* bq1 = (const float*)d_in[11];
    const float* Wk1 = (const float*)d_in[12];  const float* bk1 = (const float*)d_in[13];
    const float* Wv  = (const float*)d_in[14];  const float* bv  = (const float*)d_in[15];
    const float* Wo  = (const float*)d_in[16];  const float* bo  = (const float*)d_in[17];
    const float* gq  = (const float*)d_in[18];  const float* dq  = (const float*)d_in[19];
    const float* gk  = (const float*)d_in[20];  const float* dk  = (const float*)d_in[21];
    const float* gq1 = (const float*)d_in[22];  const float* dq1 = (const float*)d_in[23];
    const float* gk1 = (const float*)d_in[24];  const float* dk1 = (const float*)d_in[25];
    const float* gv  = (const float*)d_in[26];  const float* dv  = (const float*)d_in[27];
    const float* gp  = (const float*)d_in[28];  const float* dp  = (const float*)d_in[29];
    float* out = (float*)d_out;

    const size_t ACT   = (size_t)4096 * 1024;
    const size_t WELEM = (size_t)1024 * 1024;
    char* ws = (char*)d_ws;
    auto take = [&](size_t bytes) {
        void* p = ws;
        ws += (bytes + 255) & ~(size_t)255;
        return p;
    };
    u16* aQ  = (u16*)take(ACT * 2);  u16* aK  = (u16*)take(ACT * 2);
    u16* aQ1 = (u16*)take(ACT * 2);  u16* aK1 = (u16*)take(ACT * 2);
    u16* aV  = (u16*)take(ACT * 2);
    u16* wQ  = (u16*)take(WELEM * 2); u16* wK  = (u16*)take(WELEM * 2);
    u16* wQ1 = (u16*)take(WELEM * 2); u16* wK1 = (u16*)take(WELEM * 2);
    u16* wV  = (u16*)take(WELEM * 2); u16* wO  = (u16*)take(WELEM * 2);
    u16* pQ  = (u16*)take(ACT * 2);  u16* pK  = (u16*)take(ACT * 2);
    u16* pQ1 = (u16*)take(ACT * 2);  u16* pK1 = (u16*)take(ACT * 2);
    u16* pV  = (u16*)take(ACT * 2);
    float* ctx0 = (float*)take(ACT * 4);
    float* ctx1 = (float*)take(ACT * 4);
    u16* cln = (u16*)take(ACT * 2);

    // 1) weights fp32 -> bf16
    {
        const int blocks = (int)(WELEM / 1024);
        k_cvt_bf16<<<blocks, 256, 0, stream>>>(Wq,  wQ,  (int)WELEM);
        k_cvt_bf16<<<blocks, 256, 0, stream>>>(Wk,  wK,  (int)WELEM);
        k_cvt_bf16<<<blocks, 256, 0, stream>>>(Wq1, wQ1, (int)WELEM);
        k_cvt_bf16<<<blocks, 256, 0, stream>>>(Wk1, wK1, (int)WELEM);
        k_cvt_bf16<<<blocks, 256, 0, stream>>>(Wv,  wV,  (int)WELEM);
        k_cvt_bf16<<<blocks, 256, 0, stream>>>(Wo,  wO,  (int)WELEM);
    }

    // 2) LayerNorm -> bf16
    k_layernorm_bf16<<<4096, 256, 0, stream>>>(x_q,  nullptr, gq,  dq,  aQ);
    k_layernorm_bf16<<<4096, 256, 0, stream>>>(x_k,  nullptr, gk,  dk,  aK);
    k_layernorm_bf16<<<4096, 256, 0, stream>>>(x_q1, nullptr, gq1, dq1, aQ1);
    k_layernorm_bf16<<<4096, 256, 0, stream>>>(x_k1, nullptr, gk1, dk1, aK1);
    k_layernorm_bf16<<<4096, 256, 0, stream>>>(x_v,  nullptr, gv,  dv,  aV);

    // 3) projections (Y = X W^T + b) -> bf16
    {
        dim3 grid(8, 32);   // N/128, M/128
        k_gemm_wmma<0><<<grid, 256, 0, stream>>>(aQ,  wQ,  bq,  nullptr, nullptr, pQ);
        k_gemm_wmma<0><<<grid, 256, 0, stream>>>(aK,  wK,  bk,  nullptr, nullptr, pK);
        k_gemm_wmma<0><<<grid, 256, 0, stream>>>(aQ1, wQ1, bq1, nullptr, nullptr, pQ1);
        k_gemm_wmma<0><<<grid, 256, 0, stream>>>(aK1, wK1, bk1, nullptr, nullptr, pK1);
        k_gemm_wmma<0><<<grid, 256, 0, stream>>>(aV,  wV,  bv,  nullptr, nullptr, pV);
    }

    // 4) dual flash attention (streams write separate f32 ctx buffers)
    {
        dim3 grid(16, 32);  // S/128, B*NH
        k_attention<<<grid, 256, 0, stream>>>(pQ,  pK,  pV, amask, ctx0);
        k_attention<<<grid, 256, 0, stream>>>(pQ1, pK1, pV, amask, ctx1);
    }

    // 5) post-LN of (ctx0 + ctx1) -> bf16
    k_layernorm_bf16<<<4096, 256, 0, stream>>>(ctx0, ctx1, gp, dp, cln);

    // 6) output projection + bias + residual(value tensor) -> f32 d_out
    {
        dim3 grid(8, 32);
        k_gemm_wmma<1><<<grid, 256, 0, stream>>>(cln, wO, bo, x_v, out, nullptr);
    }
}
